// GATv2Layer4View_86208583566034
// MI455X (gfx1250) — matrix operand order repaired and checked by reading, not compiled, not run on hardware
//
#include <hip/hip_runtime.h>
#include <math.h>

typedef __attribute__((ext_vector_type(16))) _Float16 v16h;
typedef __attribute__((ext_vector_type(8)))  _Float16 v8h;
typedef __attribute__((ext_vector_type(4)))  _Float16 v4h;
typedef __attribute__((ext_vector_type(8)))  float    v8f;

#define HEADS_ 4
#define F_OUT_ 32
#define B_     2
#define V_     4
#define N_     10000
#define FIN_   64
#define D_     128
#define E_     160000
#define EP_    170000
#define BV_    8
#define ROWS_  80000          /* BV_*N_ == B_*N_*V_ */
#define NEG_SLOPE_ 0.2f

// ---------------------------------------------------------------- fragments
// A fragment 16x32 f16 read from an LDS tile sA[16][ldk] (f16):
//   M = lane%16 ; element e -> K = (e/8)*16 + (lane/16)*8 + (e%8)
//   => elements 0..7  : 8 contiguous halfs at K = k_base      + 8*(lane/16)
//      elements 8..15 : 8 contiguous halfs at K = k_base + 16 + 8*(lane/16)
__device__ __forceinline__ v16h lds_a16x32(const _Float16* __restrict__ sA, int ldk,
                                           int k_base, int lane) {
  const _Float16* p = sA + (lane & 15) * ldk + k_base + ((lane >> 4) << 3);
  union { v16h v; v8h h[2]; } u;
  u.h[0] = *(const v8h*)p;          // ds_load_b128
  u.h[1] = *(const v8h*)(p + 16);   // ds_load_b128
  return u.v;
}

// B fragment 32x16 f16 from pre-converted row-major f16 weight Wh[j,k]:
//   N = lane%16 ; element e -> K = (e/8)*16 + (lane/16)*8 + (e%8)
__device__ __forceinline__ v16h load_b32x16_h(const _Float16* __restrict__ Wh, int ld,
                                              int n_base, int k_base, int lane) {
  const _Float16* col = Wh + (size_t)(n_base + (lane & 15)) * ld + k_base + ((lane >> 4) << 3);
  union { v16h v; v8h h[2]; } u;
  u.h[0] = *(const v8h*)col;        // global_load_b128
  u.h[1] = *(const v8h*)(col + 16); // global_load_b128
  return u.v;
}

__device__ __forceinline__ v4h cvt4(float4 f) {
  v4h h;
  h[0] = (_Float16)f.x; h[1] = (_Float16)f.y;
  h[2] = (_Float16)f.z; h[3] = (_Float16)f.w;
  return h;
}

__device__ __forceinline__ unsigned fmap(float f) {
  unsigned b = __float_as_uint(f);
  return (b & 0x80000000u) ? ~b : (b | 0x80000000u);
}
__device__ __forceinline__ float funmap(unsigned u) {
  return (u & 0x80000000u) ? __uint_as_float(u ^ 0x80000000u) : __uint_as_float(~u);
}

// async lane-wise 16B copy global -> LDS (GLOBAL_LOAD_ASYNC_TO_LDS_B128, ASYNCcnt)
__device__ __forceinline__ void async_g2l_b128(const float* g, const float* lds_generic) {
  unsigned lds = (unsigned)(size_t)lds_generic;   // low 32 bits = LDS offset
  unsigned long long ga = (unsigned long long)(size_t)g;
  asm volatile("global_load_async_to_lds_b128 %0, %1, off"
               :: "v"(lds), "v"(ga) : "memory");
}
__device__ __forceinline__ void wait_asynccnt0() {
  asm volatile("s_wait_asynccnt 0" ::: "memory");
}

// ---------------------------------------------------------------- kernels
__global__ void k_zero(float* __restrict__ p, long n) {
  long i = (long)blockIdx.x * blockDim.x + threadIdx.x;
  long s = (long)gridDim.x * blockDim.x;
  for (; i < n; i += s) p[i] = 0.f;
}

__global__ void k_cvt16(const float* __restrict__ src, _Float16* __restrict__ dst, int n) {
  int i = blockIdx.x * 256 + threadIdx.x;
  if (i < n) dst[i] = (_Float16)src[i];
}

// h[m, d] = x[m, :] . W[d, :]    m in [0,80000), d in [0,128)
__global__ void k_gemm_h(const float* __restrict__ x, const _Float16* __restrict__ Wh,
                         float* __restrict__ h_buf) {
  __shared__ _Float16 sA[16 * FIN_];
  int lane = threadIdx.x & 31;
  int wave = threadIdx.x >> 5;          // 8 waves -> 8 N-tiles cover D=128
  int m0 = blockIdx.x * 16;
  // cooperative stage: tile is 16*64 = 1024 contiguous floats of x
  {
    int t = threadIdx.x;                // 4 floats per thread
    float4 f = *(const float4*)(x + (size_t)m0 * FIN_ + t * 4);
    *(v4h*)&sA[t * 4] = cvt4(f);
  }
  __syncthreads();
  int n_base = wave * 16;
  v8f c = {};
#pragma unroll
  for (int kb = 0; kb < FIN_; kb += 32) {
    v16h a = lds_a16x32(sA, FIN_, kb, lane);
    v16h b = load_b32x16_h(Wh, FIN_, n_base, kb, lane);
    c = __builtin_amdgcn_wmma_f32_16x16x32_f16(false, a, false, b, (short)0, c, false, false);
  }
  int nn = n_base + (lane & 15);
  int mrow = m0 + 8 * (lane >> 4);
#pragma unroll
  for (int r = 0; r < 8; ++r)
    h_buf[(size_t)(mrow + r) * D_ + nn] = c[r];
}

// per-node attention scores: s_src/s_dst[bv*N+n, hh]
__global__ void k_node_scores(const float* __restrict__ h_buf, const float* __restrict__ att,
                              float* __restrict__ s_src, float* __restrict__ s_dst) {
  int idx = blockIdx.x * blockDim.x + threadIdx.x;
  if (idx >= BV_ * N_ * HEADS_) return;
  int hh = idx & 3;
  int node = idx >> 2;                       // bv*N + n
  const float* hp = h_buf + (size_t)node * D_ + hh * F_OUT_;
  const float* as = att + hh * (2 * F_OUT_);
  const float* ad = as + F_OUT_;
  float ss = 0.f, sd = 0.f;
#pragma unroll
  for (int f = 0; f < F_OUT_; f += 4) {
    float4 hv = *(const float4*)(hp + f);
    float4 av = *(const float4*)(as + f);
    float4 dv = *(const float4*)(ad + f);
    float h0 = hv.x >= 0.f ? hv.x : NEG_SLOPE_ * hv.x;
    float h1 = hv.y >= 0.f ? hv.y : NEG_SLOPE_ * hv.y;
    float h2 = hv.z >= 0.f ? hv.z : NEG_SLOPE_ * hv.z;
    float h3 = hv.w >= 0.f ? hv.w : NEG_SLOPE_ * hv.w;
    ss += h0 * av.x + h1 * av.y + h2 * av.z + h3 * av.w;
    sd += h0 * dv.x + h1 * dv.y + h2 * dv.z + h3 * dv.w;
  }
  s_src[idx] = ss;
  s_dst[idx] = sd;
}

// per-edge scores (mean over BV) + global per-head max
__global__ void k_edge_scores(const int* __restrict__ ei, const float* __restrict__ s_src,
                              const float* __restrict__ s_dst, float* __restrict__ escore,
                              unsigned* __restrict__ gmax) {
  __shared__ float red[4 * 256];
  int e = blockIdx.x * 256 + threadIdx.x;
  float sc[4] = {-3.0e38f, -3.0e38f, -3.0e38f, -3.0e38f};
  if (e < EP_) {
    int s = (e < E_) ? ei[e] : (e - E_);
    int d = (e < E_) ? ei[E_ + e] : (e - E_);
    float acc[4] = {0.f, 0.f, 0.f, 0.f};
#pragma unroll
    for (int bv = 0; bv < BV_; ++bv) {
      float4 a = *(const float4*)(s_src + (size_t)(bv * N_ + s) * 4);
      float4 b = *(const float4*)(s_dst + (size_t)(bv * N_ + d) * 4);
      acc[0] += a.x + b.x; acc[1] += a.y + b.y;
      acc[2] += a.z + b.z; acc[3] += a.w + b.w;
    }
#pragma unroll
    for (int hh = 0; hh < 4; ++hh) {
      sc[hh] = acc[hh] * 0.125f;
      escore[(size_t)e * 4 + hh] = sc[hh];
    }
  }
#pragma unroll
  for (int hh = 0; hh < 4; ++hh) red[hh * 256 + threadIdx.x] = sc[hh];
  __syncthreads();
  for (int s2 = 128; s2 > 0; s2 >>= 1) {
    if ((int)threadIdx.x < s2)
#pragma unroll
      for (int hh = 0; hh < 4; ++hh)
        red[hh * 256 + threadIdx.x] =
            fmaxf(red[hh * 256 + threadIdx.x], red[hh * 256 + threadIdx.x + s2]);
    __syncthreads();
  }
  if (threadIdx.x == 0)
#pragma unroll
    for (int hh = 0; hh < 4; ++hh) atomicMax(&gmax[hh], fmap(red[hh * 256]));
}

// exp(score - max) in place + global per-head sum
__global__ void k_exp_sum(float* __restrict__ escore, const unsigned* __restrict__ gmax,
                          float* __restrict__ gsum) {
  __shared__ float red[4 * 256];
  int e = blockIdx.x * 256 + threadIdx.x;
  float ex[4] = {0.f, 0.f, 0.f, 0.f};
  if (e < EP_) {
#pragma unroll
    for (int hh = 0; hh < 4; ++hh) {
      float m = funmap(gmax[hh]);
      float v = __expf(escore[(size_t)e * 4 + hh] - m);
      escore[(size_t)e * 4 + hh] = v;
      ex[hh] = v;
    }
  }
#pragma unroll
  for (int hh = 0; hh < 4; ++hh) red[hh * 256 + threadIdx.x] = ex[hh];
  __syncthreads();
  for (int s2 = 128; s2 > 0; s2 >>= 1) {
    if ((int)threadIdx.x < s2)
#pragma unroll
      for (int hh = 0; hh < 4; ++hh)
        red[hh * 256 + threadIdx.x] += red[hh * 256 + threadIdx.x + s2];
    __syncthreads();
  }
  if (threadIdx.x == 0)
#pragma unroll
    for (int hh = 0; hh < 4; ++hh) atomicAdd(&gsum[hh], red[hh * 256]);
}

// scatter-add: out_gat[bv, dst, :] += h[bv, src, :] * w[e, head]
__global__ void k_scatter(const int* __restrict__ ei, const float* __restrict__ h_buf,
                          const float* __restrict__ escore, const float* __restrict__ gsum,
                          float* __restrict__ out_gat) {
  int e = blockIdx.x;
  int c = threadIdx.x;                      // 0..127
  int s = (e < E_) ? ei[e] : (e - E_);
  int d = (e < E_) ? ei[E_ + e] : (e - E_);
  int hh = c >> 5;
  float w = escore[(size_t)e * 4 + hh] / gsum[hh];
#pragma unroll
  for (int bv = 0; bv < BV_; ++bv) {
    float val = h_buf[((size_t)(bv * N_ + s)) * D_ + c] * w;
    atomicAdd(&out_gat[((size_t)(bv * N_ + d)) * D_ + c], val);
  }
}

// qkv[r, j] = q_in[r, :] . in_proj_w[j, :] + in_proj_b[j]; r = (b*N+n)*V+v
__global__ void k_gemm_qkv(const float* __restrict__ out_gat, const _Float16* __restrict__ ipwh,
                           const float* __restrict__ ipb, float* __restrict__ qkv) {
  __shared__ _Float16 sA[16 * D_];
  int lane = threadIdx.x & 31;
  int wave = threadIdx.x >> 5;
  int m0 = blockIdx.x * 16;
  // cooperative stage: 16 rows x 128 cols, rows permuted via (b,n,v) -> (bv,n)
  {
    int idx = threadIdx.x * 8;        // 8 halfs per thread
    int row = idx >> 7;
    int col = idx & 127;
    int r = m0 + row;
    int bb = r / (N_ * V_);
    int rem = r - bb * (N_ * V_);
    int n = rem >> 2;
    int v = rem & 3;
    const float* g = out_gat + ((size_t)((bb * V_ + v) * N_ + n)) * D_ + col;
    float4 f0 = *(const float4*)g;
    float4 f1 = *(const float4*)(g + 4);
    *(v4h*)&sA[idx] = cvt4(f0);
    *(v4h*)&sA[idx + 4] = cvt4(f1);
  }
  __syncthreads();
  int nb0 = wave * 16, nb1 = (wave + 8) * 16, nb2 = (wave + 16) * 16;  // 24 N-tiles
  v8f c0 = {}, c1 = {}, c2 = {};
#pragma unroll
  for (int kb = 0; kb < D_; kb += 32) {
    v16h a = lds_a16x32(sA, D_, kb, lane);
    v16h b0 = load_b32x16_h(ipwh, D_, nb0, kb, lane);
    c0 = __builtin_amdgcn_wmma_f32_16x16x32_f16(false, a, false, b0, (short)0, c0, false, false);
    v16h b1 = load_b32x16_h(ipwh, D_, nb1, kb, lane);
    c1 = __builtin_amdgcn_wmma_f32_16x16x32_f16(false, a, false, b1, (short)0, c1, false, false);
    v16h b2 = load_b32x16_h(ipwh, D_, nb2, kb, lane);
    c2 = __builtin_amdgcn_wmma_f32_16x16x32_f16(false, a, false, b2, (short)0, c2, false, false);
  }
  int mrow = m0 + 8 * (lane >> 4);
  int ln = lane & 15;
#pragma unroll
  for (int r8 = 0; r8 < 8; ++r8) {
    size_t row = (size_t)(mrow + r8);
    qkv[row * 384 + nb0 + ln] = c0[r8] + ipb[nb0 + ln];
    qkv[row * 384 + nb1 + ln] = c1[r8] + ipb[nb1 + ln];
    qkv[row * 384 + nb2 + ln] = c2[r8] + ipb[nb2 + ln];
  }
}

// per-node MHA over V=4 tokens; o[r, d]. qkv rows staged via async load-to-LDS.
__global__ void k_attn(const float* __restrict__ qkv, float* __restrict__ o) {
  __shared__ float sq[4 * 384];             // this node's q|k|v rows (6 KB)
  __shared__ float attn[64];
  int node = blockIdx.x;                    // b*N + n
  int t = threadIdx.x;                      // 0..63
  const float* gsrc = qkv + (size_t)node * 4 * 384;
#pragma unroll
  for (int i = 0; i < 6; ++i) {             // 6 * 64 * 4 = 1536 floats
    int off = (i * 64 + t) * 4;
    async_g2l_b128(gsrc + off, sq + off);
  }
  wait_asynccnt0();
  __syncthreads();
  {
    int h = t >> 4, q = (t >> 2) & 3, k = t & 3;
    const float* qp = sq + q * 384 + h * F_OUT_;
    const float* kp = sq + k * 384 + D_ + h * F_OUT_;
    float acc = 0.f;
#pragma unroll
    for (int i = 0; i < F_OUT_; i += 4) {
      float4 a = *(const float4*)(qp + i);
      float4 b = *(const float4*)(kp + i);
      acc += a.x * b.x + a.y * b.y + a.z * b.z + a.w * b.w;
    }
    attn[t] = acc * 0.17677669529663687f;   // 1/sqrt(32)
  }
  __syncthreads();
  if (t < 16) {                              // t = h*4 + q
    int base = t * 4;
    float m = fmaxf(fmaxf(attn[base], attn[base + 1]), fmaxf(attn[base + 2], attn[base + 3]));
    float e0 = __expf(attn[base] - m), e1 = __expf(attn[base + 1] - m);
    float e2 = __expf(attn[base + 2] - m), e3 = __expf(attn[base + 3] - m);
    float inv = 1.f / (e0 + e1 + e2 + e3);
    attn[base] = e0 * inv; attn[base + 1] = e1 * inv;
    attn[base + 2] = e2 * inv; attn[base + 3] = e3 * inv;
  }
  __syncthreads();
#pragma unroll
  for (int i = 0; i < 8; ++i) {
    int idx = t * 8 + i;                     // 0..511 = 4 rows x 128 cols
    int vq = idx >> 7;
    int d = idx & 127;
    int h2 = d >> 5;
    float acc = 0.f;
#pragma unroll
    for (int k = 0; k < 4; ++k)
      acc += attn[(h2 * 4 + vq) * 4 + k] * sq[k * 384 + 256 + d];
    o[((size_t)(node * 4 + vq)) * D_ + d] = acc;
  }
}

// out[b,v,n,j] = o[r,:].out_proj_w[j,:] + out_proj_b[j] + bias[j]; r=(b*N+n)*V+v
__global__ void k_gemm_out(const float* __restrict__ o, const _Float16* __restrict__ opwh,
                           const float* __restrict__ opb, const float* __restrict__ bias,
                           float* __restrict__ out) {
  __shared__ _Float16 sA[16 * D_];
  int lane = threadIdx.x & 31;
  int wave = threadIdx.x >> 5;
  int m0 = blockIdx.x * 16;
  // cooperative stage: tile is 16*128 = 2048 contiguous floats of o
  {
    int idx = threadIdx.x * 8;
    const float* g = o + (size_t)m0 * D_ + idx;
    float4 f0 = *(const float4*)g;
    float4 f1 = *(const float4*)(g + 4);
    *(v4h*)&sA[idx] = cvt4(f0);
    *(v4h*)&sA[idx + 4] = cvt4(f1);
  }
  __syncthreads();
  int n_base = wave * 16;
  v8f c = {};
#pragma unroll
  for (int kb = 0; kb < D_; kb += 32) {
    v16h a = lds_a16x32(sA, D_, kb, lane);
    v16h b = load_b32x16_h(opwh, D_, n_base, kb, lane);
    c = __builtin_amdgcn_wmma_f32_16x16x32_f16(false, a, false, b, (short)0, c, false, false);
  }
  int j = n_base + (lane & 15);
  float add = opb[j] + bias[j];
  int mrow = m0 + 8 * (lane >> 4);
#pragma unroll
  for (int r8 = 0; r8 < 8; ++r8) {
    int row = mrow + r8;                     // (b*N+n)*V + v
    int bb = row / (N_ * V_);
    int rem = row - bb * (N_ * V_);
    int n = rem >> 2;
    int v = rem & 3;
    out[((size_t)((bb * V_ + v) * N_ + n)) * D_ + j] = c[r8] + add;
  }
}

// ---------------------------------------------------------------- launch
extern "C" void kernel_launch(void* const* d_in, const int* in_sizes, int n_in,
                              void* d_out, int out_size, void* d_ws, size_t ws_size,
                              hipStream_t stream) {
  (void)in_sizes; (void)n_in; (void)out_size; (void)ws_size;
  const float* x    = (const float*)d_in[0];
  const float* W    = (const float*)d_in[1];
  const float* att  = (const float*)d_in[2];
  const float* ipw  = (const float*)d_in[3];
  const float* ipb  = (const float*)d_in[4];
  const float* opw  = (const float*)d_in[5];
  const float* opb  = (const float*)d_in[6];
  const float* bias = (const float*)d_in[7];
  const int*   ei   = (const int*)d_in[8];
  float* out = (float*)d_out;

  float* ws       = (float*)d_ws;
  float* h_buf    = ws;                          // 10,240,000 (reused as o after scatter)
  float* out_gat  = h_buf + 10240000;            // 10,240,000
  float* s_src    = out_gat + 10240000;          //    320,000
  float* s_dst    = s_src + 320000;              //    320,000
  float* escore   = s_dst + 320000;              //    680,000
  unsigned* gmax  = (unsigned*)(escore + 680000);//          4
  float* gsum     = (float*)(gmax + 4);          //          4
  float* qkv      = gsum + 4;                    // 30,720,000
  _Float16* Wh    = (_Float16*)(qkv + 30720000); //      8,192 halfs
  _Float16* ipwh  = Wh + 8192;                   //     49,152 halfs
  _Float16* opwh  = ipwh + 49152;                //     16,384 halfs

  k_zero<<<2048, 256, 0, stream>>>(out_gat, 10240000L);
  k_zero<<<1, 32, 0, stream>>>((float*)gmax, 8L);
  k_cvt16<<<(8192 + 255) / 256, 256, 0, stream>>>(W, Wh, 8192);
  k_cvt16<<<(49152 + 255) / 256, 256, 0, stream>>>(ipw, ipwh, 49152);
  k_cvt16<<<(16384 + 255) / 256, 256, 0, stream>>>(opw, opwh, 16384);

  k_gemm_h<<<ROWS_ / 16, 256, 0, stream>>>(x, Wh, h_buf);
  k_node_scores<<<(BV_ * N_ * HEADS_ + 255) / 256, 256, 0, stream>>>(h_buf, att, s_src, s_dst);
  k_edge_scores<<<(EP_ + 255) / 256, 256, 0, stream>>>(ei, s_src, s_dst, escore, gmax);
  k_exp_sum<<<(EP_ + 255) / 256, 256, 0, stream>>>(escore, gmax, gsum);
  k_scatter<<<EP_, 128, 0, stream>>>(ei, h_buf, escore, gsum, out_gat);
  k_gemm_qkv<<<ROWS_ / 16, 256, 0, stream>>>(out_gat, ipwh, ipb, qkv);
  k_attn<<<B_ * N_, 64, 0, stream>>>(qkv, h_buf);           // h_buf reused as o
  k_gemm_out<<<ROWS_ / 16, 256, 0, stream>>>(h_buf, opwh, opb, bias, out);
}
